// EncoderModel_17978733101214
// MI455X (gfx1250) — compile-verified, matrix operand order checked by reading, not run on hardware
//
#include <hip/hip_runtime.h>
#include <hip/hip_bf16.h>
#include <cstdint>
#include <math.h>

// Problem constants (reference: B,T,V,E,H,L = 128,512,128,256,1024,2)
#define BB   128
#define TT   512
#define VV   128
#define EE   256
#define HH   1024
#define G4H  4096
#define NWG  32          // persistent workgroups; WG w owns h-columns [32w,32w+32)
#define TPB  256         // 8 waves of 32
#define K0S  40          // (E+H)/32 k-steps for layer0
#define K1S  64          // 2H/32 k-steps for layer1

typedef __bf16 v16bf __attribute__((ext_vector_type(16)));
typedef float  v8f   __attribute__((ext_vector_type(8)));

// packed activation buffer sizes (elements of bf16), per parity buffer
#define A0X_ELEMS (8 * 8 * 512)    // x part:  8 m-tiles x 8  ksteps x (32 lanes x 16)
#define A0H_ELEMS (8 * 32 * 512)   // h0 part: 8 m-tiles x 32 ksteps
#define A1_ELEMS  (8 * 64 * 512)   // [nh0 | h1]: 8 m-tiles x 64 ksteps

__device__ __forceinline__ float sigf(float x) { return 1.0f / (1.0f + __expf(-x)); }
__device__ __forceinline__ float tanhf_(float x) { return 2.0f * sigf(2.0f * x) - 1.0f; }

// one k-step of the gate GEMM: 1 A-frag vs 8 B-frags; loads grouped into one
// clause, then 8 back-to-back WMMAs. Pointers are pre-offset by lane; callers
// must NOT unroll the k loop so live B-fragments stay bounded (no spills).
__device__ __forceinline__ void gemm_kstep(const __bf16* __restrict__ ap,
                                           const __bf16* __restrict__ wp,
                                           int kstride_elems, v8f acc[8]) {
  v16bf af = *(const v16bf*)ap;
  v16bf bfr[8];
#pragma unroll
  for (int nt = 0; nt < 8; ++nt)
    bfr[nt] = *(const v16bf*)(wp + (size_t)nt * kstride_elems);
#pragma unroll
  for (int nt = 0; nt < 8; ++nt)
    acc[nt] = __builtin_amdgcn_wmma_f32_16x16x32_bf16(
        false, af, false, bfr[nt], (short)0, acc[nt], false, false);
}

// ---------------- grid-wide software barrier (persistent kernel) -------------
__device__ __forceinline__ void grid_barrier(unsigned* bar, unsigned* phase) {
  __threadfence();
  __syncthreads();
  if (threadIdx.x == 0) {
    unsigned target = ++(*phase);
    unsigned prev = __hip_atomic_fetch_add(&bar[0], 1u, __ATOMIC_ACQ_REL,
                                           __HIP_MEMORY_SCOPE_AGENT);
    if (prev == NWG - 1u) {
      __hip_atomic_store(&bar[0], 0u, __ATOMIC_RELAXED, __HIP_MEMORY_SCOPE_AGENT);
      __hip_atomic_store(&bar[1], target, __ATOMIC_RELEASE, __HIP_MEMORY_SCOPE_AGENT);
    } else {
      while (__hip_atomic_load(&bar[1], __ATOMIC_ACQUIRE,
                               __HIP_MEMORY_SCOPE_AGENT) < target) {
        __builtin_amdgcn_s_sleep(2);
      }
    }
  }
  __syncthreads();
  __threadfence();
}

// ---------------- setup kernels ---------------------------------------------
__global__ void zero_ws_kernel(uint4* p, unsigned n16) {
  uint4 z; z.x = z.y = z.z = z.w = 0u;
  for (unsigned i = blockIdx.x * blockDim.x + threadIdx.x; i < n16;
       i += gridDim.x * blockDim.x)
    p[i] = z;
}

__global__ void pack_emb_kernel(const float* __restrict__ ef,
                                __bf16* __restrict__ eb, int n) {
  int i = blockIdx.x * blockDim.x + threadIdx.x;
  if (i < n) eb[i] = (__bf16)ef[i];
}

// Pack fp32 weights [K x 4096] into per-WG WMMA B-fragment tiles (bf16):
//   Wp[((wg*8 + nt)*Ksteps + k)*512 + lane*16 + i]
// lane&15 = N column within 16-wide tile; K interleave matches 16-bit WMMA
// operand layout: lanes<16 hold K {0-7,16-23}, lanes>=16 hold K {8-15,24-31}.
__global__ void pack_w_kernel(const float* __restrict__ W,
                              __bf16* __restrict__ Wp, int Ksteps, long total) {
  long id = (long)blockIdx.x * blockDim.x + threadIdx.x;
  if (id >= total) return;
  int i = (int)(id & 15);
  int l = (int)((id >> 4) & 31);
  long rest = id >> 9;
  int k  = (int)(rest % Ksteps); rest /= Ksteps;
  int nt = (int)(rest & 7);
  int wg = (int)(rest >> 3);
  int g = nt >> 1, cti = nt & 1;
  int n  = g * HH + wg * 32 + cti * 16 + (l & 15);
  int kk = k * 32 + ((l & 16) ? 8 : 0) + ((i & 7) | ((i & 8) << 1));
  Wp[id] = (__bf16)W[(size_t)kk * G4H + n];
}

// ---------------- persistent 2-layer LSTM -----------------------------------
__global__ void __launch_bounds__(TPB, 1)
lstm_persistent(const int* __restrict__ batch, const int* __restrict__ lens,
                const __bf16* __restrict__ emb,
                const __bf16* __restrict__ W0p, const __bf16* __restrict__ W1p,
                const float* __restrict__ b0, const float* __restrict__ b1,
                __bf16* __restrict__ A0x, __bf16* __restrict__ A0h,
                __bf16* __restrict__ A1, unsigned* __restrict__ bar,
                float* __restrict__ out) {
  const int wg   = blockIdx.x;
  const int wave = threadIdx.x >> 5;
  const int lane = threadIdx.x & 31;
  const int hi8  = (lane & 16) ? 8 : 0;  // C/D layout: lanes>=16 hold rows M+8
  const int nlo  = lane & 15;            // C/D + B layout: N column in tile
  const int row0 = wave * 16 + hi8;      // this lane's first batch row

  int lenr[8];
#pragma unroll
  for (int r = 0; r < 8; ++r) lenr[r] = lens[row0 + r];

  float bias0[8], bias1[8];
#pragma unroll
  for (int nt = 0; nt < 8; ++nt) {
    int n = (nt >> 1) * HH + wg * 32 + (nt & 1) * 16 + nlo;
    bias0[nt] = b0[n];
    bias1[nt] = b1[n];
  }

  // recurrent state lives in registers for the whole sequence
  float c0r[2][8], h0r[2][8], c1r[2][8], h1r[2][8];
#pragma unroll
  for (int c = 0; c < 2; ++c)
#pragma unroll
    for (int r = 0; r < 8; ++r) {
      c0r[c][r] = 0.f; h0r[c][r] = 0.f; c1r[c][r] = 0.f; h1r[c][r] = 0.f;
    }

  unsigned phase = 0;

  for (int t = 0; t < TT; ++t) {
    // ---- S1: gather x_t = emb[batch[:,t]] into A-fragment layout -----------
    {
      __bf16* a0x_w = A0x + (size_t)(t & 1) * A0X_ELEMS;
#pragma unroll
      for (int ti = 0; ti < 2; ++ti) {
        int tile = wg + ti * NWG;  // 0..63 : m-tile(8) x kstep(8)
        int m = tile >> 3, k = tile & 7;
#pragma unroll
        for (int e = 0; e < 2; ++e) {
          int idx = (threadIdx.x << 1) | e;  // 0..511 within tile
          int l = idx >> 4, i = idx & 15;
          int row = m * 16 + (l & 15);
          int kk  = k * 32 + ((l & 16) ? 8 : 0) + ((i & 7) | ((i & 8) << 1));
          int tok = batch[row * TT + t];
          a0x_w[tile * 512 + idx] = emb[tok * EE + kk];
        }
      }
    }
    grid_barrier(bar, &phase);

    // ---- S2: layer-0 gates GEMM + cell update ------------------------------
    {
      const __bf16* ap = A0x + (size_t)(t & 1) * A0X_ELEMS +
                         (size_t)wave * 8 * 512 + lane * 16;
      const __bf16* wp = W0p + (size_t)wg * 8 * K0S * 512 + lane * 16;
      v8f acc[8];
#pragma unroll
      for (int nt = 0; nt < 8; ++nt)
#pragma unroll
        for (int r = 0; r < 8; ++r) acc[nt][r] = bias0[nt];

      // x part: ksteps 0..7 (rolled; running pointers -> incremental addr)
#pragma unroll 1
      for (int k = 0; k < 8; ++k) {
        __builtin_prefetch(wp + 512, 0, 1);
        gemm_kstep(ap, wp, K0S * 512, acc);
        ap += 512;
        wp += 512;
      }
      // h0 part: ksteps 8..39
      ap = A0h + (size_t)(t & 1) * A0H_ELEMS + (size_t)wave * 32 * 512 +
           lane * 16;
#pragma unroll 1
      for (int k = 8; k < K0S; ++k) {
        __builtin_prefetch(wp + 512, 0, 1);
        gemm_kstep(ap, wp, K0S * 512, acc);
        ap += 512;
        wp += 512;
      }

      __bf16* a0h_w = A0h + (size_t)((t + 1) & 1) * A0H_ELEMS;  // h0 for t+1
      __bf16* a1_w  = A1  + (size_t)(t & 1) * A1_ELEMS;         // nh0 for layer1
#pragma unroll
      for (int cti = 0; cti < 2; ++cti) {
#pragma unroll
        for (int r = 0; r < 8; ++r) {
          float gi = acc[0 * 2 + cti][r], gj = acc[1 * 2 + cti][r];
          float gf = acc[2 * 2 + cti][r], go = acc[3 * 2 + cti][r];
          float nc = c0r[cti][r] * sigf(gf + 1.0f) + sigf(gi) * tanhf_(gj);
          float nh = tanhf_(nc) * sigf(go);
          bool msk = (t < lenr[r]);
          nc = msk ? nc : c0r[cti][r];
          nh = msk ? nh : h0r[cti][r];
          c0r[cti][r] = nc;
          h0r[cti][r] = nh;
          // scatter into A-fragment layout (this WG's 32 cols == one kstep)
          int ko = cti * 16 + nlo;          // K offset within 32-chunk
          int row = row0 + r;
          int lp = (row & 15) | ((ko & 8) << 1);
          int ip = (ko & 7) | ((ko & 16) >> 1);
          __bf16 hb = (__bf16)nh;
          a0h_w[(size_t)(wave * 32 + wg) * 512 + lp * 16 + ip] = hb;
          a1_w [(size_t)(wave * 64 + wg) * 512 + lp * 16 + ip] = hb;
        }
      }
    }
    grid_barrier(bar, &phase);

    // ---- S3: layer-1 gates GEMM + cell update ------------------------------
    {
      const __bf16* ap = A1 + (size_t)(t & 1) * A1_ELEMS +
                         (size_t)wave * 64 * 512 + lane * 16;
      const __bf16* wp = W1p + (size_t)wg * 8 * K1S * 512 + lane * 16;
      v8f acc[8];
#pragma unroll
      for (int nt = 0; nt < 8; ++nt)
#pragma unroll
        for (int r = 0; r < 8; ++r) acc[nt][r] = bias1[nt];

#pragma unroll 1
      for (int k = 0; k < K1S; ++k) {
        __builtin_prefetch(wp + 512, 0, 1);
        gemm_kstep(ap, wp, K1S * 512, acc);
        ap += 512;
        wp += 512;
      }

      __bf16* a1h_w = A1 + (size_t)((t + 1) & 1) * A1_ELEMS;  // h1 for t+1
#pragma unroll
      for (int cti = 0; cti < 2; ++cti) {
#pragma unroll
        for (int r = 0; r < 8; ++r) {
          float gi = acc[0 * 2 + cti][r], gj = acc[1 * 2 + cti][r];
          float gf = acc[2 * 2 + cti][r], go = acc[3 * 2 + cti][r];
          float nc = c1r[cti][r] * sigf(gf + 1.0f) + sigf(gi) * tanhf_(gj);
          float nh = tanhf_(nc) * sigf(go);
          bool msk = (t < lenr[r]);
          nc = msk ? nc : c1r[cti][r];
          nh = msk ? nh : h1r[cti][r];
          c1r[cti][r] = nc;
          h1r[cti][r] = nh;
          int ko = cti * 16 + nlo;
          int row = row0 + r;
          int lp = (row & 15) | ((ko & 8) << 1);
          int ip = (ko & 7) | ((ko & 16) >> 1);
          a1h_w[(size_t)(wave * 64 + (32 + wg)) * 512 + lp * 16 + ip] = (__bf16)nh;
        }
      }
    }
    // no barrier needed here: next step's S1->S2 barrier rendezvous covers it
    // (all buffers touched across the boundary are parity double-buffered)
  }

  // h1 registers froze at t = len-1 per row == encoder_last_output
#pragma unroll
  for (int cti = 0; cti < 2; ++cti)
#pragma unroll
    for (int r = 0; r < 8; ++r) {
      int row = row0 + r;
      int col = wg * 32 + cti * 16 + nlo;
      out[(size_t)row * HH + col] = h1r[cti][r];
    }
}

// ---------------- host entry -------------------------------------------------
extern "C" void kernel_launch(void* const* d_in, const int* in_sizes, int n_in,
                              void* d_out, int out_size, void* d_ws, size_t ws_size,
                              hipStream_t stream) {
  const int*   batch = (const int*)d_in[0];
  const int*   lens  = (const int*)d_in[1];
  const float* embf  = (const float*)d_in[2];
  const float* W0    = (const float*)d_in[3];
  const float* b0    = (const float*)d_in[4];
  const float* W1    = (const float*)d_in[5];
  const float* b1    = (const float*)d_in[6];
  float* out = (float*)d_out;

  char* ws = (char*)d_ws;
  size_t off = 0;
  auto take = [&](size_t bytes) {
    size_t o = off;
    off = (off + bytes + 255) & ~(size_t)255;
    return o;
  };
  size_t o_bar = take(256);
  size_t o_A0x = take((size_t)2 * A0X_ELEMS * 2);
  size_t o_A0h = take((size_t)2 * A0H_ELEMS * 2);
  size_t o_A1  = take((size_t)2 * A1_ELEMS * 2);
  size_t zero_bytes = off;  // barrier + all activation state zeroed each call
  size_t o_emb = take((size_t)VV * EE * 2);
  size_t o_W0p = take((size_t)(EE + HH) * G4H * 2);
  size_t o_W1p = take((size_t)(2 * HH) * G4H * 2);
  (void)ws_size; (void)in_sizes; (void)n_in; (void)out_size;

  unsigned nz16 = (unsigned)(zero_bytes / 16);
  hipLaunchKernelGGL(zero_ws_kernel, dim3(512), dim3(256), 0, stream,
                     (uint4*)ws, nz16);

  hipLaunchKernelGGL(pack_emb_kernel, dim3((VV * EE + 255) / 256), dim3(256), 0,
                     stream, embf, (__bf16*)(ws + o_emb), VV * EE);

  long w0tot = (long)NWG * 8 * K0S * 512;
  hipLaunchKernelGGL(pack_w_kernel, dim3((unsigned)((w0tot + 255) / 256)),
                     dim3(256), 0, stream, W0, (__bf16*)(ws + o_W0p), K0S, w0tot);
  long w1tot = (long)NWG * 8 * K1S * 512;
  hipLaunchKernelGGL(pack_w_kernel, dim3((unsigned)((w1tot + 255) / 256)),
                     dim3(256), 0, stream, W1, (__bf16*)(ws + o_W1p), K1S, w1tot);

  hipLaunchKernelGGL(lstm_persistent, dim3(NWG), dim3(TPB), 0, stream,
                     batch, lens, (const __bf16*)(ws + o_emb),
                     (const __bf16*)(ws + o_W0p), (const __bf16*)(ws + o_W1p),
                     b0, b1,
                     (__bf16*)(ws + o_A0x), (__bf16*)(ws + o_A0h),
                     (__bf16*)(ws + o_A1), (unsigned*)(ws + o_bar), out);
}
